// GAT_Modelv5_Regression_37993280700730
// MI455X (gfx1250) — compile-verified, matrix operand order checked by reading, not run on hardware
//
#include <hip/hip_runtime.h>
#include <hip/hip_bf16.h>

// ---------------- problem constants (from reference) ----------------
constexpr int NN   = 50000;
constexpr int EE   = 600000;
constexpr int GG   = 256;
constexpr int FIN  = 128;
constexpr int HH   = 3;     // heads
constexpr int CC   = 64;    // channels per head
constexpr int HC   = 192;   // H*C
constexpr int EP   = EE + NN;   // edges + self loops = 650000

typedef __attribute__((ext_vector_type(16))) __bf16 v16bf;
typedef __attribute__((ext_vector_type(8)))  float  v8f;

// ---------------- helpers ----------------
__device__ __forceinline__ void atomicMaxFloat(float* addr, float val) {
    if (val >= 0.0f) atomicMax((int*)addr, __float_as_int(val));
    else             atomicMin((unsigned int*)addr, (unsigned int)__float_as_int(val));
}

__device__ __forceinline__ unsigned pack2bf(float x, float y) {
    union { __bf16 b[2]; unsigned u; } t;
    t.b[0] = (__bf16)x; t.b[1] = (__bf16)y;
    return t.u;
}

__global__ void fill_f32(float* __restrict__ p, float v, int n) {
    int i = blockIdx.x * blockDim.x + threadIdx.x;
    if (i < n) p[i] = v;
}

// ---------------- edge list with self loops ----------------
__global__ void build_edges(const int* __restrict__ edge_index,
                            int* __restrict__ src, int* __restrict__ dst) {
    int i = blockIdx.x * blockDim.x + threadIdx.x;
    if (i < EE)      { src[i] = edge_index[i]; dst[i] = edge_index[EE + i]; }
    else if (i < EP) { src[i] = i - EE;        dst[i] = i - EE; }
}

// ---------------- operand packing into WMMA fragment layout ----------------
// A fragment (16x32 bf16, ISA 7.12.2): lane row = lane&15; dword p holds K pair
//   k = (p&3)*2 + (p>>2)*16 + (lane>>4)*8.
// Packed A: [rowtile][kstep][lane][8 dwords]  (32B per lane, contiguous)
__global__ void pack_a(const float* __restrict__ X, unsigned* __restrict__ Apk,
                       int rows, int K) {
    const int KB = K >> 5;                         // k-steps
    const int nfrag = (rows >> 4) * KB;
    int idx = blockIdx.x * blockDim.x + threadIdx.x;
    if (idx >= nfrag * 32) return;
    const int lane = idx & 31;
    const int f    = idx >> 5;
    const int ki   = f % KB;
    const int rt   = f / KB;
    const int row  = (rt << 4) + (lane & 15);
    const int kb   = ki << 5;
    const int hofs = (lane >> 4) * 8;
    const float* __restrict__ xrow = X + (size_t)row * K + kb;
    unsigned* __restrict__ o = Apk + (size_t)idx * 8;
#pragma unroll
    for (int p = 0; p < 8; ++p) {
        const int k = (p & 3) * 2 + (p >> 2) * 16 + hofs;
        o[p] = pack2bf(xrow[k], xrow[k + 1]);
    }
}

// B fragment (32x16 bf16): lane col = lane&15; dword p holds K pair (same k map).
// Packed B: [kstep][coltile][lane][8 dwords]
__global__ void pack_b(const float* __restrict__ W, unsigned* __restrict__ Bpk,
                       int K, int M) {
    const int MT = M >> 4;                         // col tiles
    const int KB = K >> 5;
    const int nfrag = KB * MT;
    int idx = blockIdx.x * blockDim.x + threadIdx.x;
    if (idx >= nfrag * 32) return;
    const int lane = idx & 31;
    const int f    = idx >> 5;
    const int jt   = f % MT;
    const int ki   = f / MT;
    const int col  = (jt << 4) + (lane & 15);
    const int kb   = ki << 5;
    const int hofs = (lane >> 4) * 8;
    unsigned* __restrict__ o = Bpk + (size_t)idx * 8;
#pragma unroll
    for (int p = 0; p < 8; ++p) {
        const int k = kb + (p & 3) * 2 + (p >> 2) * 16 + hofs;
        o[p] = pack2bf(W[(size_t)k * M + col], W[(size_t)(k + 1) * M + col]);
    }
}

// ---------------- WMMA GEMM on packed operands ----------------
// One wave computes a 16x64 macro-tile: 1 A frag x 4 B frags -> 4 wmma/k-step.
// C/D layout: lane -> col = lane&15, VGPR v -> row = v + 8*(lane>>4).
template <bool BIAS, bool RELU>
__global__ void wmma_gemm_pk(const unsigned* __restrict__ Apk,
                             const unsigned* __restrict__ Bpk,
                             float* __restrict__ Y, int K, int M,
                             const float* __restrict__ bias) {
    const int lane = threadIdx.x;          // 0..31
    const int half = lane >> 4;
    const int lidx = lane & 15;
    const int rt   = blockIdx.x;           // row tile (16 rows)
    const int jt0  = blockIdx.y * 4;       // first of 4 col tiles
    const int KB   = K >> 5;
    const int MT   = M >> 4;

    v8f acc[4] = {};
    const unsigned* __restrict__ ap = Apk + ((size_t)rt * KB * 32 + lane) * 8;
    for (int ki = 0; ki < KB; ++ki) {
        const v16bf a = *(const v16bf*)(ap + (size_t)ki * 32 * 8);
#pragma unroll
        for (int j = 0; j < 4; ++j) {
            const v16bf b = *(const v16bf*)(
                Bpk + (((size_t)ki * MT + jt0 + j) * 32 + lane) * 8);
            acc[j] = __builtin_amdgcn_wmma_f32_16x16x32_bf16(
                         false, a, false, b, (short)0, acc[j], false, false);
        }
    }

#pragma unroll
    for (int j = 0; j < 4; ++j) {
        const int c = ((jt0 + j) << 4) + lidx;
        float badd = 0.0f;
        if (BIAS) badd = bias[c];
#pragma unroll
        for (int v = 0; v < 8; ++v) {
            const int r = (rt << 4) + v + half * 8;
            float val = acc[j][v];
            if (BIAS) val += badd;
            if (RELU) val = fmaxf(val, 0.0f);
            Y[(size_t)r * M + c] = val;
        }
    }
}

// ---------------- GATv2 edge phase ----------------
__global__ void edge_logits_kernel(const int* __restrict__ src, const int* __restrict__ dst,
                                   const float* __restrict__ xl, const float* __restrict__ xr,
                                   const float* __restrict__ att,
                                   float* __restrict__ logits, float* __restrict__ mmax) {
    int idx = blockIdx.x * blockDim.x + threadIdx.x;
    if (idx >= EP * HH) return;
    const int h = idx / EP;
    const int e = idx - h * EP;
    const int s = src[e], d = dst[e];
    const float* pl = xl + (size_t)s * HC + h * CC;
    const float* pr = xr + (size_t)d * HC + h * CC;
    const float* pa = att + h * CC;
    float acc = 0.0f;
#pragma unroll 8
    for (int c = 0; c < CC; ++c) {
        float v = pl[c] + pr[c];
        v = (v > 0.0f) ? v : 0.2f * v;       // leaky_relu(0.2)
        acc = fmaf(v, pa[c], acc);
    }
    logits[(size_t)h * EP + e] = acc;
    atomicMaxFloat(&mmax[(size_t)h * NN + d], acc);
}

__global__ void edge_exp_kernel(const int* __restrict__ dst,
                                const float* __restrict__ logits, const float* __restrict__ mmax,
                                float* __restrict__ alpha, float* __restrict__ denom) {
    int idx = blockIdx.x * blockDim.x + threadIdx.x;
    if (idx >= EP * HH) return;
    const int h = idx / EP;
    const int e = idx - h * EP;
    const int d = dst[e];
    const float a = __expf(logits[(size_t)h * EP + e] - mmax[(size_t)h * NN + d]);
    alpha[(size_t)h * EP + e] = a;
    atomicAdd(&denom[(size_t)h * NN + d], a);
}

__global__ void edge_aggregate_kernel(const int* __restrict__ src, const int* __restrict__ dst,
                                      const float* __restrict__ xl,
                                      const float* __restrict__ alpha, const float* __restrict__ denom,
                                      float* __restrict__ agg) {
    int idx = blockIdx.x * blockDim.x + threadIdx.x;
    if (idx >= EP * HH) return;
    const int h = idx / EP;
    const int e = idx - h * EP;
    const int s = src[e], d = dst[e];
    const float w = alpha[(size_t)h * EP + e] / (denom[(size_t)h * NN + d] + 1e-16f);
    const float* pl = xl + (size_t)s * HC + h * CC;
    float* po = agg + (size_t)d * HC + h * CC;
#pragma unroll 4
    for (int c = 0; c < CC; ++c)
        atomicAdd(&po[c], pl[c] * w);
}

// ---------------- conv finalize ----------------
__global__ void conv_finalize_concat(const float* __restrict__ agg, const float* __restrict__ bias,
                                     float* __restrict__ out) {
    int idx = blockIdx.x * blockDim.x + threadIdx.x;
    if (idx >= NN * HC) return;
    out[idx] = agg[idx] + bias[idx % HC];
}

__global__ void conv_finalize_mean(const float* __restrict__ agg, const float* __restrict__ bias,
                                   float* __restrict__ out) {
    int idx = blockIdx.x * blockDim.x + threadIdx.x;
    if (idx >= NN * CC) return;
    const int node = idx / CC;
    const int c    = idx - node * CC;
    const float* p = agg + (size_t)node * HC + c;
    out[idx] = (p[0] + p[CC] + p[2 * CC]) * (1.0f / 3.0f) + bias[c];
}

// ---------------- batch norm ----------------
__global__ void bn_stats(const float* __restrict__ x, float* __restrict__ sum,
                         float* __restrict__ sumsq, int Cn) {
    const int c = threadIdx.x;              // blockDim.x == Cn
    float s = 0.0f, s2 = 0.0f;
    for (int r = blockIdx.x; r < NN; r += gridDim.x) {
        const float v = x[(size_t)r * Cn + c];
        s += v; s2 = fmaf(v, v, s2);
    }
    atomicAdd(&sum[c], s);
    atomicAdd(&sumsq[c], s2);
}

template <bool RESID>
__global__ void bn_apply(float* __restrict__ x, const float* __restrict__ sum,
                         const float* __restrict__ sumsq,
                         const float* __restrict__ gamma, const float* __restrict__ beta,
                         const float* __restrict__ residual, int Cn) {
    int idx = blockIdx.x * blockDim.x + threadIdx.x;
    if (idx >= NN * Cn) return;
    const int c = idx % Cn;
    const float inv_n = 1.0f / (float)NN;
    const float mean  = sum[c] * inv_n;
    const float var   = sumsq[c] * inv_n - mean * mean;
    float v = (x[idx] - mean) * rsqrtf(var + 1e-5f) * gamma[c] + beta[c];
    v = fmaxf(v, 0.0f);
    if (RESID) v += residual[idx];
    x[idx] = v;
}

// ---------------- graph readout (mean, std, max) ----------------
__global__ void readout_accum(const float* __restrict__ out4, const int* __restrict__ batch,
                              float* __restrict__ s1, float* __restrict__ s2,
                              float* __restrict__ mx, float* __restrict__ cnt) {
    int idx = blockIdx.x * blockDim.x + threadIdx.x;
    if (idx >= NN * CC) return;
    const int node = idx / CC;
    const int c    = idx - node * CC;
    const int g    = batch[node];
    const float v  = out4[idx];
    atomicAdd(&s1[(size_t)g * CC + c], v);
    atomicAdd(&s2[(size_t)g * CC + c], v * v);
    atomicMaxFloat(&mx[(size_t)g * CC + c], v);   // post-relu values >= 0; init 0 matches ref
    if (c == 0) atomicAdd(&cnt[g], 1.0f);
}

__global__ void readout_finalize(const float* __restrict__ s1, const float* __restrict__ s2,
                                 const float* __restrict__ mx, const float* __restrict__ cnt,
                                 float* __restrict__ h) {
    int idx = blockIdx.x * blockDim.x + threadIdx.x;
    if (idx >= GG * CC) return;
    const int g = idx / CC;
    const int c = idx - g * CC;
    const float n    = fmaxf(cnt[g], 1.0f);
    const float mean = s1[idx] / n;
    const float var  = s2[idx] / n - mean * mean;
    const float sd   = sqrtf(fmaxf(var, 0.0f) + 1e-5f);
    const float m    = (cnt[g] > 0.0f) ? mx[idx] : 0.0f;
    h[(size_t)g * HC + c]            = mean;
    h[(size_t)g * HC + CC + c]       = sd;
    h[(size_t)g * HC + 2 * CC + c]   = m;
}

// ---------------- fc3: [G,256] @ [256,1] + b ----------------
__global__ void fc3_kernel(const float* __restrict__ h2, const float* __restrict__ W,
                           const float* __restrict__ b, float* __restrict__ out) {
    const int g = blockIdx.x;
    const int lane = threadIdx.x;
    float s = 0.0f;
    for (int k = lane; k < 256; k += 32)
        s = fmaf(h2[(size_t)g * 256 + k], W[k], s);
#pragma unroll
    for (int off = 16; off > 0; off >>= 1) s += __shfl_down(s, off, 32);
    if (lane == 0) out[g] = s + b[0];
}

// ---------------- host orchestration ----------------
static inline int cdiv(long a, long b) { return (int)((a + b - 1) / b); }

extern "C" void kernel_launch(void* const* d_in, const int* in_sizes, int n_in,
                              void* d_out, int out_size, void* d_ws, size_t ws_size,
                              hipStream_t stream) {
    (void)in_sizes; (void)n_in; (void)out_size; (void)ws_size;
    const float* x          = (const float*)d_in[0];
    const int*   edge_index = (const int*)  d_in[1];
    const int*   batch      = (const int*)  d_in[2];

    // ---- workspace carve-out ----
    char* ws = (char*)d_ws;
    size_t off = 0;
    auto carve = [&](size_t bytes) -> void* {
        void* p = ws + off;
        off += (bytes + 255) & ~(size_t)255;
        return p;
    };
    int*      src    = (int*)     carve((size_t)EP * 4);
    int*      dst    = (int*)     carve((size_t)EP * 4);
    float*    xl     = (float*)   carve((size_t)NN * HC * 4);
    float*    xr     = (float*)   carve((size_t)NN * HC * 4);
    float*    F0     = (float*)   carve((size_t)NN * HC * 4);   // out1 (residual)
    float*    F1     = (float*)   carve((size_t)NN * HC * 4);   // out2, later out4 [N,64]
    float*    F2     = (float*)   carve((size_t)NN * HC * 4);   // out3
    float*    agg    = (float*)   carve((size_t)NN * HC * 4);
    float*    logits = (float*)   carve((size_t)EP * HH * 4);
    float*    alpha  = (float*)   carve((size_t)EP * HH * 4);
    float*    mbuf   = (float*)   carve((size_t)NN * HH * 4);
    float*    denom  = (float*)   carve((size_t)NN * HH * 4);
    float*    bnsum  = (float*)   carve((size_t)HC * 4);
    float*    bnsq   = (float*)   carve((size_t)HC * 4);
    float*    cnt    = (float*)   carve((size_t)GG * 4);
    float*    r1     = (float*)   carve((size_t)GG * CC * 4);
    float*    r2     = (float*)   carve((size_t)GG * CC * 4);
    float*    rmx    = (float*)   carve((size_t)GG * CC * 4);
    float*    hbuf   = (float*)   carve((size_t)GG * HC * 4);
    float*    h1     = (float*)   carve((size_t)GG * 256 * 4);
    float*    h2     = (float*)   carve((size_t)GG * 256 * 4);
    unsigned* Apk    = (unsigned*)carve((size_t)NN * HC * 2);   // packed bf16 A (max K=256 rowsxK)
    unsigned* BpkL   = (unsigned*)carve((size_t)256 * 256 * 2); // packed bf16 B (Wl / fc)
    unsigned* BpkR   = (unsigned*)carve((size_t)256 * 256 * 2); // packed bf16 B (Wr)

    const dim3 b256(256);

    // edge list (+ self loops)
    build_edges<<<cdiv(EP, 256), b256, 0, stream>>>(edge_index, src, dst);

    const float* cur = x;
    int K = FIN;
    float* dests[4] = { F0, F1, F2, F1 /* out4 reuses F1 as [N,64] */ };

    for (int l = 0; l < 4; ++l) {
        const int pi = 3 + l * 6;
        const float* Wl    = (const float*)d_in[pi + 0];
        const float* Wr    = (const float*)d_in[pi + 1];
        const float* att   = (const float*)d_in[pi + 2];
        const float* bias  = (const float*)d_in[pi + 3];
        const float* gamma = (const float*)d_in[pi + 4];
        const float* beta  = (const float*)d_in[pi + 5];
        const bool concat  = (l < 3);
        const int  Cn      = concat ? HC : CC;
        float* dest        = dests[l];

        // pack operands into WMMA fragment layout (bf16)
        const int naf = (NN / 16) * (K / 32) * 32;
        pack_a<<<cdiv(naf, 256), b256, 0, stream>>>(cur, Apk, NN, K);
        const int nbf = (K / 32) * (HC / 16) * 32;
        pack_b<<<cdiv(nbf, 256), b256, 0, stream>>>(Wl, BpkL, K, HC);
        pack_b<<<cdiv(nbf, 256), b256, 0, stream>>>(Wr, BpkR, K, HC);

        // linear projections via WMMA (16x64 macro-tile per wave)
        wmma_gemm_pk<false, false><<<dim3(NN / 16, HC / 64), dim3(32), 0, stream>>>(
            Apk, BpkL, xl, K, HC, nullptr);
        wmma_gemm_pk<false, false><<<dim3(NN / 16, HC / 64), dim3(32), 0, stream>>>(
            Apk, BpkR, xr, K, HC, nullptr);

        // init per-layer accumulators
        fill_f32<<<cdiv(NN * HH, 256), b256, 0, stream>>>(mbuf, -__builtin_inff(), NN * HH);
        fill_f32<<<cdiv(NN * HH, 256), b256, 0, stream>>>(denom, 0.0f, NN * HH);
        fill_f32<<<cdiv((long)NN * HC, 256), b256, 0, stream>>>(agg, 0.0f, NN * HC);
        fill_f32<<<1, dim3(HC), 0, stream>>>(bnsum, 0.0f, HC);
        fill_f32<<<1, dim3(HC), 0, stream>>>(bnsq, 0.0f, HC);

        // attention softmax over destination nodes (3 passes, L2-resident)
        const int egrid = cdiv((long)EP * HH, 256);
        edge_logits_kernel<<<egrid, b256, 0, stream>>>(src, dst, xl, xr, att, logits, mbuf);
        edge_exp_kernel<<<egrid, b256, 0, stream>>>(dst, logits, mbuf, alpha, denom);
        edge_aggregate_kernel<<<egrid, b256, 0, stream>>>(src, dst, xl, alpha, denom, agg);

        // bias + head combine
        if (concat) conv_finalize_concat<<<cdiv((long)NN * HC, 256), b256, 0, stream>>>(agg, bias, dest);
        else        conv_finalize_mean<<<cdiv((long)NN * CC, 256), b256, 0, stream>>>(agg, bias, dest);

        // batch norm + relu (+ residual after relu on layer 3)
        bn_stats<<<512, dim3(Cn), 0, stream>>>(dest, bnsum, bnsq, Cn);
        if (l == 2)
            bn_apply<true><<<cdiv((long)NN * Cn, 256), b256, 0, stream>>>(
                dest, bnsum, bnsq, gamma, beta, F0, Cn);
        else
            bn_apply<false><<<cdiv((long)NN * Cn, 256), b256, 0, stream>>>(
                dest, bnsum, bnsq, gamma, beta, nullptr, Cn);

        cur = dest;
        K = HC;
    }

    // ---- readout: mean / std / max per graph ----
    fill_f32<<<1, b256, 0, stream>>>(cnt, 0.0f, GG);
    fill_f32<<<cdiv(GG * CC, 256), b256, 0, stream>>>(r1, 0.0f, GG * CC);
    fill_f32<<<cdiv(GG * CC, 256), b256, 0, stream>>>(r2, 0.0f, GG * CC);
    fill_f32<<<cdiv(GG * CC, 256), b256, 0, stream>>>(rmx, 0.0f, GG * CC);
    readout_accum<<<cdiv((long)NN * CC, 256), b256, 0, stream>>>(F1, batch, r1, r2, rmx, cnt);
    readout_finalize<<<cdiv(GG * CC, 256), b256, 0, stream>>>(r1, r2, rmx, cnt, hbuf);

    // ---- MLP head (same packed-WMMA path, fused bias+relu) ----
    const float* W1 = (const float*)d_in[27]; const float* b1 = (const float*)d_in[28];
    const float* W2 = (const float*)d_in[29]; const float* b2 = (const float*)d_in[30];
    const float* W3 = (const float*)d_in[31]; const float* b3 = (const float*)d_in[32];

    pack_a<<<cdiv((GG / 16) * (HC / 32) * 32, 256), b256, 0, stream>>>(hbuf, Apk, GG, HC);
    pack_b<<<cdiv((HC / 32) * (256 / 16) * 32, 256), b256, 0, stream>>>(W1, BpkL, HC, 256);
    wmma_gemm_pk<true, true><<<dim3(GG / 16, 256 / 64), dim3(32), 0, stream>>>(
        Apk, BpkL, h1, HC, 256, b1);

    pack_a<<<cdiv((GG / 16) * (256 / 32) * 32, 256), b256, 0, stream>>>(h1, Apk, GG, 256);
    pack_b<<<cdiv((256 / 32) * (256 / 16) * 32, 256), b256, 0, stream>>>(W2, BpkL, 256, 256);
    wmma_gemm_pk<true, true><<<dim3(GG / 16, 256 / 64), dim3(32), 0, stream>>>(
        Apk, BpkL, h2, 256, 256, b2);

    fc3_kernel<<<GG, dim3(32), 0, stream>>>(h2, W3, b3, (float*)d_out);
}
